// ODEG_64261300683067
// MI455X (gfx1250) — matrix operand order repaired and checked by reading, not compiled, not run on hardware
//
#include <hip/hip_runtime.h>
#include <math.h>

#define BN 32
#define NN 512
#define TT 12
#define FF 64
#define TF 768   /* T*F */
#define HIDN 128
#define STEPS 12
#define NJB 8    /* j-tile blocks per batch in transpose (NN/64) */

typedef __bf16 bf16_t;
typedef bf16_t v16bf __attribute__((ext_vector_type(16)));
typedef float  v8f   __attribute__((ext_vector_type(8)));

struct U32x8 { uint4 lo, hi; };

__device__ __forceinline__ unsigned short f2bf(float x) {
    unsigned int u = __builtin_bit_cast(unsigned int, x);
    u += 0x7FFFu + ((u >> 16) & 1u);          // round-to-nearest-even
    return (unsigned short)(u >> 16);
}

__device__ __forceinline__ v16bf ld_frag(const unsigned short* p0,
                                         const unsigned short* p1) {
    U32x8 t;
    t.lo = *(const uint4*)p0;
    t.hi = *(const uint4*)p1;
    return __builtin_bit_cast(v16bf, t);
}

// async copy: 32 bytes global -> LDS (2 x b128), ASYNCcnt-tracked.
// INST_OFFSET is added to BOTH the global and LDS address (ISA 8.4), so the
// second b128 reuses the same register operands with offset:16.
__device__ __forceinline__ void async_ld32(unsigned ldsOff,
                                           const unsigned short* g) {
    unsigned long long ga = (unsigned long long)(uintptr_t)g;
    asm volatile("global_load_async_to_lds_b128 %0, %1, off"
                 :: "v"(ldsOff), "v"(ga) : "memory");
    asm volatile("global_load_async_to_lds_b128 %0, %1, off offset:16"
                 :: "v"(ldsOff), "v"(ga) : "memory");
}

// ---------------- adj f32 -> bf16 (once per launch) ----------------
__global__ void cvt_adj_kernel(const float* __restrict__ adj,
                               unsigned short* __restrict__ adjb, int n) {
    int i = blockIdx.x * blockDim.x + threadIdx.x;
    if (i < n) adjb[i] = f2bf(adj[i]);
}

// --- stage f32 [B,N,TF] -> bf16 transposed hbT [B,TF,N] + column sums ---
// part[b][jblk][c] = sum over the 64 j's of this tile of stage[b,j,c]
__global__ __launch_bounds__(256) void cvt_transpose_kernel(
        const float* __restrict__ stage, unsigned short* __restrict__ hbT,
        float* __restrict__ part) {
    __shared__ unsigned short tile[64][65];
    __shared__ float ps[4][64];
    int b = blockIdx.z, jb = blockIdx.y, j0 = jb * 64, c0 = blockIdx.x * 64;
    int t = threadIdx.x, lc = t & 63, lr = t >> 6;
    const float* src = stage + ((size_t)b * NN + j0) * TF + c0;
    float csum = 0.f;
    #pragma unroll
    for (int rep = 0; rep < 16; ++rep) {
        int jr = rep * 4 + lr;
        float v = src[(size_t)jr * TF + lc];
        tile[jr][lc] = f2bf(v);
        csum += v;
    }
    ps[lr][lc] = csum;
    __syncthreads();
    unsigned short* dst = hbT + ((size_t)b * TF + c0) * NN + j0;
    #pragma unroll
    for (int rep = 0; rep < 16; ++rep) {
        int cr = rep * 4 + lr;
        dst[(size_t)cr * NN + lc] = tile[lc][cr];
    }
    if (lr == 0)
        part[((size_t)b * NJB + jb) * TF + c0 + lc] =
            ps[0][lc] + ps[1][lc] + ps[2][lc] + ps[3][lc];
}

// ---- fold partials -> mean -> MLP -> escale[b,f] = exp(mod) ----
__global__ __launch_bounds__(256) void mlp_kernel(
        const float* __restrict__ part,
        const float* __restrict__ mw1, const float* __restrict__ mb1,
        const float* __restrict__ mw2, const float* __restrict__ mb2,
        const float* __restrict__ alpha, float* __restrict__ escale) {
    __shared__ float red[256];
    __shared__ float xm[FF];
    __shared__ float hid[HIDN];
    int b = blockIdx.x, t = threadIdx.x, f = t & 63, sub = t >> 6;
    float v = 0.f;
    #pragma unroll
    for (int jb = 0; jb < NJB; jb += 4) {
        const float* p = part + ((size_t)b * NJB + jb + sub) * TF + f;
        #pragma unroll
        for (int tt = 0; tt < TT; ++tt) v += p[tt * FF];
    }
    red[t] = v;
    __syncthreads();
    if (sub == 0)
        xm[f] = (red[f] + red[f + 64] + red[f + 128] + red[f + 192]) *
                (1.0f / (float)(NN * TT));
    __syncthreads();
    if (t < HIDN) {
        float a = mb1[t];
        #pragma unroll 8
        for (int q = 0; q < FF; ++q) a += xm[q] * mw1[q * HIDN + t];
        hid[t] = fmaxf(a, 0.f);
    }
    __syncthreads();
    if (t < FF) {
        float a = mb2[t];
        #pragma unroll 8
        for (int q = 0; q < HIDN; ++q) a += hid[q] * mw2[q * FF + t];
        float mod = tanhf(alpha[t] + a) * 0.5f;
        escale[b * FF + t] = expf(mod);
    }
}

// ---------------- main WMMA GEMM + fused RK4 epilogue --------------
// C[i,c] = sum_j adj[i,j]*h[b,j,c]; val = (C - h)*escale
// S==0: acc = wk*val ; next = h0 + ak*val
// S==1/2: acc += wk*val ; next = h0 + ak*val
// S==3: hout = h0 + acc + wk*val
template <int S>
__global__ __launch_bounds__(256) void gemm_rk4_kernel(
        const unsigned short* __restrict__ adjb,
        const unsigned short* __restrict__ hbT,
        const float* __restrict__ stage,
        const float* __restrict__ escale,
        const float* __restrict__ h0,
        float* __restrict__ accb,
        float* __restrict__ nextStage,
        float* __restrict__ hout,
        float wk, float ak) {
    __shared__ __align__(16) unsigned short As[2][128][40];
    __shared__ __align__(16) unsigned short Bs[2][128][40];
    int b = blockIdx.z;
    int m0 = blockIdx.y * 128;
    int c0 = blockIdx.x * 128;
    int t = threadIdx.x;
    int lane = t & 31, wave = t >> 5;
    int wm = wave & 1, wn = wave >> 1;       // 2 x 4 wave grid -> 64x32 / wave
    int lrow = t >> 1, lhalf = (t & 1) * 16; // 32B per thread per matrix row

    const unsigned short* Ag = adjb + (size_t)(m0 + lrow) * NN + lhalf;
    const unsigned short* Bg = hbT + ((size_t)b * TF + c0 + lrow) * NN + lhalf;

    // per-thread LDS byte offsets for async destinations (double buffered)
    unsigned offA[2], offB[2];
    offA[0] = (unsigned)(uintptr_t)&As[0][lrow][lhalf];
    offA[1] = (unsigned)(uintptr_t)&As[1][lrow][lhalf];
    offB[0] = (unsigned)(uintptr_t)&Bs[0][lrow][lhalf];
    offB[1] = (unsigned)(uintptr_t)&Bs[1][lrow][lhalf];

    v8f acc[4][2];
    v8f zero = {0.f, 0.f, 0.f, 0.f, 0.f, 0.f, 0.f, 0.f};
    #pragma unroll
    for (int i = 0; i < 4; ++i)
        #pragma unroll
        for (int j = 0; j < 2; ++j) acc[i][j] = zero;

    // prologue: stage chunk 0 into buffer 0 (4 async b128 per thread)
    async_ld32(offA[0], Ag);
    async_ld32(offB[0], Bg);

    int q = lane >> 4;    // lane half -> K sub-range
    int lm = lane & 15;

    #pragma unroll
    for (int k = 0; k < 16; ++k) {
        const int buf = k & 1;
        if (k < 15) {
            const int nb = buf ^ 1;
            async_ld32(offA[nb], Ag + (k + 1) * 32);
            async_ld32(offB[nb], Bg + (k + 1) * 32);
            asm volatile("s_wait_asynccnt 0x4" ::: "memory");
        } else {
            asm volatile("s_wait_asynccnt 0x0" ::: "memory");
        }
        __syncthreads();   // all waves: chunk-k tiles resident in LDS

        v16bf af[4];
        #pragma unroll
        for (int mt = 0; mt < 4; ++mt) {
            // A 16x32 bf16: lane half q -> K {8q..8q+7, 16+8q..16+8q+7}
            const unsigned short* pa = &As[buf][wm * 64 + mt * 16 + lm][0];
            af[mt] = ld_frag(pa + 8 * q, pa + 16 + 8 * q);
        }
        v16bf bfr[2];
        #pragma unroll
        for (int nt = 0; nt < 2; ++nt) {
            // B 32x16 bf16: lane half q -> K {8q..8q+7, 16+8q..16+8q+7}, col = lane&15
            const unsigned short* pb = &Bs[buf][wn * 32 + nt * 16 + lm][0];
            bfr[nt] = ld_frag(pb + 8 * q, pb + 16 + 8 * q);
        }
        #pragma unroll
        for (int mt = 0; mt < 4; ++mt)
            #pragma unroll
            for (int nt = 0; nt < 2; ++nt)
                acc[mt][nt] = __builtin_amdgcn_wmma_f32_16x16x32_bf16(
                    false, af[mt], false, bfr[nt], (short)0, acc[mt][nt],
                    false, false);
        __syncthreads();   // reads done before next chunk overwrites buffer
    }

    // fused RK4 epilogue; C/D layout: VGPR r -> row r + 8*(lane>=16), col lane&15
    int lh = lane >> 4;
    float es[2];
    int cg0[2];
    #pragma unroll
    for (int nt = 0; nt < 2; ++nt) {
        cg0[nt] = c0 + wn * 32 + nt * 16 + lm;
        es[nt] = escale[b * FF + (cg0[nt] & 63)];
    }
    #pragma unroll
    for (int mt = 0; mt < 4; ++mt) {
        #pragma unroll
        for (int r = 0; r < 8; ++r) {
            int ig = m0 + wm * 64 + mt * 16 + lh * 8 + r;
            size_t rowb = ((size_t)b * NN + ig) * TF;
            #pragma unroll
            for (int nt = 0; nt < 2; ++nt) {
                size_t idx = rowb + cg0[nt];
                float val = (acc[mt][nt][r] - stage[idx]) * es[nt];
                float wv = wk * val;
                if constexpr (S == 0) {
                    accb[idx] = wv;
                    nextStage[idx] = h0[idx] + ak * val;
                } else if constexpr (S == 1 || S == 2) {
                    accb[idx] += wv;
                    nextStage[idx] = h0[idx] + ak * val;
                } else {
                    hout[idx] = h0[idx] + accb[idx] + wv;
                }
            }
        }
    }
}

extern "C" void kernel_launch(void* const* d_in, const int* in_sizes, int n_in,
                              void* d_out, int out_size, void* d_ws, size_t ws_size,
                              hipStream_t stream) {
    (void)in_sizes; (void)n_in; (void)out_size; (void)ws_size;
    const float* x     = (const float*)d_in[0];
    const float* adj   = (const float*)d_in[1];
    const float* mw1   = (const float*)d_in[2];
    const float* mb1   = (const float*)d_in[3];
    const float* mw2   = (const float*)d_in[4];
    const float* mb2   = (const float*)d_in[5];
    const float* alpha = (const float*)d_in[6];
    float* h = (float*)d_out;

    char* ws = (char*)d_ws;
    const size_t nElem = (size_t)BN * NN * TF;         // 12,582,912
    float*          stage  = (float*)ws;
    float*          accb   = (float*)(ws + nElem * 4);
    unsigned short* hbT    = (unsigned short*)(ws + nElem * 8);
    unsigned short* adjb   = (unsigned short*)(ws + nElem * 8 + nElem * 2);
    float*          escale = (float*)(ws + nElem * 8 + nElem * 2 +
                                      (size_t)NN * NN * 2);
    float*          part   = escale + (size_t)BN * FF;  // [BN][NJB][TF]

    hipMemcpyAsync(h, x, nElem * sizeof(float), hipMemcpyDeviceToDevice, stream);
    cvt_adj_kernel<<<(NN * NN + 255) / 256, 256, 0, stream>>>(adj, adjb, NN * NN);

    const float delta = 1.0f / 12.0f;
    const float wk[4] = {delta / 6.0f, delta / 3.0f, delta / 3.0f, delta / 6.0f};
    const float ak[4] = {0.5f * delta, 0.5f * delta, delta, 0.0f};

    dim3 gT(TF / 64, NN / 64, BN);   // transpose tiles
    dim3 gG(TF / 128, NN / 128, BN); // gemm tiles

    for (int step = 0; step < STEPS; ++step) {
        const float* src = h;
        for (int s = 0; s < 4; ++s) {
            cvt_transpose_kernel<<<gT, 256, 0, stream>>>(src, hbT, part);
            mlp_kernel<<<BN, 256, 0, stream>>>(part, mw1, mb1, mw2, mb2,
                                               alpha, escale);
            switch (s) {
                case 0:
                    gemm_rk4_kernel<0><<<gG, 256, 0, stream>>>(
                        adjb, hbT, src, escale, h, accb, stage, h, wk[0], ak[0]);
                    break;
                case 1:
                    gemm_rk4_kernel<1><<<gG, 256, 0, stream>>>(
                        adjb, hbT, src, escale, h, accb, stage, h, wk[1], ak[1]);
                    break;
                case 2:
                    gemm_rk4_kernel<2><<<gG, 256, 0, stream>>>(
                        adjb, hbT, src, escale, h, accb, stage, h, wk[2], ak[2]);
                    break;
                default:
                    gemm_rk4_kernel<3><<<gG, 256, 0, stream>>>(
                        adjb, hbT, src, escale, h, accb, nullptr, h, wk[3], ak[3]);
                    break;
            }
            src = stage;
        }
    }
}
